// ConformerBlock_8040178778628
// MI455X (gfx1250) — compile-verified, matrix operand order checked by reading, not run on hardware
//
#include <hip/hip_runtime.h>

// ---------------- problem constants ----------------
#define B_      8
#define T_      768
#define H_      512
#define NH_     8
#define DH_     64
#define SD_     128
#define INNER_  2048
#define KCONV_  31
#define PADC_   15
#define WINDOW_ 128
#define EPS_    1e-5f
#define MROWS_  (B_*T_)          // 6144
#define BTH_    (B_*T_*H_)       // 3145728

// ---------------- WMMA plumbing ----------------
typedef __attribute__((ext_vector_type(16))) __bf16 bf16x16;
typedef __attribute__((ext_vector_type(8)))  float  f32x8;

// hardware convert (v_cvt), round-to-nearest-even
__device__ __forceinline__ unsigned short f2bf(float f) {
    union { __bf16 h; unsigned short s; } c;
    c.h = (__bf16)f;
    return c.s;
}

union FragU { bf16x16 v; unsigned int u[8]; };

// Load a 16x32 bf16 operand fragment from LDS (row-major, even stride in
// shorts). Matches the CDNA5 16-bit A-matrix layout: lanes 0-15 -> rows 0-15
// with K=0..15 in VGPRs 0-3 / K=16..31 in VGPRs 4-7; lanes 16-31 get the +8
// k-half. B operands use the same loader on a transposed [n][k] LDS tile.
__device__ __forceinline__ void ld_frag(const unsigned short* base, int stride,
                                        int row0, int koff, FragU& f) {
    int lane  = threadIdx.x & 31;
    int r     = row0 + (lane & 15);
    int khalf = (lane >> 4) * 8;
    const unsigned int* p =
        (const unsigned int*)(base + (size_t)r * stride + koff);
    int d0 = khalf >> 1;
    f.u[0] = p[d0 + 0]; f.u[1] = p[d0 + 1];
    f.u[2] = p[d0 + 2]; f.u[3] = p[d0 + 3];
    f.u[4] = p[8 + d0 + 0]; f.u[5] = p[8 + d0 + 1];
    f.u[6] = p[8 + d0 + 2]; f.u[7] = p[8 + d0 + 3];
}

__device__ __forceinline__ f32x8 wmma_bf16(const FragU& a, const FragU& b, f32x8 c) {
    return __builtin_amdgcn_wmma_f32_16x16x32_bf16(
        false, a.v, false, b.v, (short)0, c, false, false);
}

// ---------------- generic WMMA GEMM ----------------
#define EPI_BIAS    0
#define EPI_SILU    1
#define EPI_RES     2
#define EPI_RESMASK 3

// C[M,N] = epilogue(A[M,K](f32) @ W[K,N] + bias), weights given PRE-TRANSPOSED
// as Bt[N][K] bf16 so both LDS tiles stage with contiguous 32B loads/stores.
// 128x128 block tile, 256 threads (8 waves, 4x2). Each wave computes 32x64 =
// 2x4 WMMA 16x16 tiles (8 f32x8 accumulators), BK=32, double-buffered LDS so
// next-slice global loads overlap the 8-WMMA burst.
__global__ __launch_bounds__(256)
void gemm_wmma(const float* __restrict__ A, const unsigned short* __restrict__ Bt,
               const float* __restrict__ bias, float* __restrict__ out,
               const float* __restrict__ resid, const unsigned char* __restrict__ kpm,
               int M, int N, int K, int mode, float scale) {
    __shared__ unsigned short As[2][128 * 34];
    __shared__ unsigned short Bs[2][128 * 34];
    int tid = threadIdx.x;
    int m0  = blockIdx.y * 128;
    int n0  = blockIdx.x * 128;
    int wv  = tid >> 5, lane = tid & 31;
    int wm  = (wv >> 1) * 32;   // 0,32,64,96
    int wn  = (wv & 1) * 64;    // 0,64

    f32x8 acc[2][4];
#pragma unroll
    for (int i = 0; i < 2; ++i)
#pragma unroll
        for (int j = 0; j < 4; ++j)
            acc[i][j] = f32x8{0.f,0.f,0.f,0.f,0.f,0.f,0.f,0.f};

    // both tiles are 128 rows x 32 cols; each thread owns a 16-elem half-row
    int r16 = tid >> 1, c16 = (tid & 1) * 16;
    const float*          aRow = A  + (size_t)(m0 + r16) * K + c16;
    const unsigned short* bRow = Bt + (size_t)(n0 + r16) * K + c16;
    unsigned short* aDst0 = &As[0][r16 * 34 + c16];
    unsigned short* bDst0 = &Bs[0][r16 * 34 + c16];
    unsigned short* aDst1 = &As[1][r16 * 34 + c16];
    unsigned short* bDst1 = &Bs[1][r16 * 34 + c16];

    {   // stage k-slice 0 into buffer 0
#pragma unroll
        for (int i = 0; i < 16; ++i) aDst0[i] = f2bf(aRow[i]);
#pragma unroll
        for (int i = 0; i < 16; ++i) bDst0[i] = bRow[i];
    }
    __syncthreads();

    int nt = K >> 5;
    for (int kt = 0; kt < nt; ++kt) {
        int cur = kt & 1;
        bool hasNext = (kt + 1) < nt;

        float          aregs[16];
        unsigned short bregs[16];
        if (hasNext) {   // issue next-slice global loads before the WMMA burst
            const float*          ag = aRow + (kt + 1) * 32;
            const unsigned short* bg = bRow + (kt + 1) * 32;
#pragma unroll
            for (int i = 0; i < 16; ++i) aregs[i] = ag[i];
#pragma unroll
            for (int i = 0; i < 16; ++i) bregs[i] = bg[i];
            if (kt + 2 < nt) {   // warm L2 two slices ahead
                __builtin_prefetch(aRow + (kt + 2) * 32, 0, 1);
                __builtin_prefetch(bRow + (kt + 2) * 32, 0, 1);
            }
        }

        const unsigned short* Ac = As[cur];
        const unsigned short* Bc = Bs[cur];
        FragU a0, a1, b0, b1, b2, b3;
        ld_frag(Ac, 34, wm,      0, a0);
        ld_frag(Ac, 34, wm + 16, 0, a1);
        ld_frag(Bc, 34, wn,      0, b0);
        ld_frag(Bc, 34, wn + 16, 0, b1);
        ld_frag(Bc, 34, wn + 32, 0, b2);
        ld_frag(Bc, 34, wn + 48, 0, b3);
        acc[0][0] = wmma_bf16(a0, b0, acc[0][0]);
        acc[0][1] = wmma_bf16(a0, b1, acc[0][1]);
        acc[0][2] = wmma_bf16(a0, b2, acc[0][2]);
        acc[0][3] = wmma_bf16(a0, b3, acc[0][3]);
        acc[1][0] = wmma_bf16(a1, b0, acc[1][0]);
        acc[1][1] = wmma_bf16(a1, b1, acc[1][1]);
        acc[1][2] = wmma_bf16(a1, b2, acc[1][2]);
        acc[1][3] = wmma_bf16(a1, b3, acc[1][3]);

        if (hasNext) {   // spill next slice to the alternate LDS buffer
            unsigned short* ad = cur ? aDst0 : aDst1;
            unsigned short* bd = cur ? bDst0 : bDst1;
#pragma unroll
            for (int i = 0; i < 16; ++i) ad[i] = f2bf(aregs[i]);
#pragma unroll
            for (int i = 0; i < 16; ++i) bd[i] = bregs[i];
        }
        __syncthreads();
    }

    int lhi = lane >> 4, llo = lane & 15;
#pragma unroll
    for (int mi = 0; mi < 2; ++mi) {
#pragma unroll
        for (int ni = 0; ni < 4; ++ni) {
            int gcol = n0 + wn + ni * 16 + llo;
            float bv = bias ? bias[gcol] : 0.f;
#pragma unroll
            for (int vr = 0; vr < 8; ++vr) {
                int grow = m0 + wm + mi * 16 + vr + lhi * 8;
                size_t oidx = (size_t)grow * N + gcol;
                float v = acc[mi][ni][vr] + bv;
                if (mode == EPI_SILU) {
                    v = v / (1.f + __expf(-v));
                } else if (mode == EPI_RES) {
                    v = resid[oidx] + scale * v;
                } else if (mode == EPI_RESMASK) {
                    int t = grow % T_;
                    int b = grow / T_;
                    if (kpm[b * T_ + t]) v = 0.f;
                    v = resid[oidx] + v;
                }
                out[oidx] = v;
            }
        }
    }
}

// ---------------- banded flash attention ----------------
// One block per (q-tile of 64, b*NH + h). 4 waves; each wave owns 16 query
// rows. Key band: [q0-128, q0+64+128) = 5 tiles of 64. Online softmax, O kept
// in WMMA accumulators.
__global__ __launch_bounds__(128)
void attn_wmma(const float* __restrict__ qkv, const unsigned char* __restrict__ kpm,
               float* __restrict__ outO) {
    __shared__ unsigned short Qs[64 * 66];
    __shared__ unsigned short Ks[64 * 66];
    __shared__ unsigned short Vt[64 * 66];   // transposed: [d][key]
    __shared__ unsigned short Ps[64 * 66];
    __shared__ float Sf[64 * 66];
    __shared__ float sM[64], sL[64], sA[64], kAdd[64];

    int tid = threadIdx.x, lane = tid & 31, wv = tid >> 5;
    int qt = blockIdx.x, bh = blockIdx.y;
    int b = bh / NH_, h = bh % NH_;
    int q0 = qt * 64;
    int wq = wv * 16;

    const float* qb = qkv + (size_t)b * T_ * (3 * H_) + h * DH_;
    const float* kb = qb + H_;
    const float* vb = qb + 2 * H_;

    {   // load Q (scaled by 1/sqrt(DH))
        int r = tid >> 1, d0 = (tid & 1) * 32;
        const float* src = qb + (size_t)(q0 + r) * (3 * H_) + d0;
        unsigned short* dst = Qs + r * 66 + d0;
#pragma unroll
        for (int i = 0; i < 32; ++i) dst[i] = f2bf(src[i] * 0.125f);
    }
    if (tid < 64) { sM[tid] = -1e30f; sL[tid] = 0.f; }

    f32x8 o[4];
#pragma unroll
    for (int i = 0; i < 4; ++i) o[i] = f32x8{0.f,0.f,0.f,0.f,0.f,0.f,0.f,0.f};
    __syncthreads();

    for (int kt = 0; kt < 5; ++kt) {
        int kbase = q0 - 128 + kt * 64;
        {   // load K tile and V tile (V transposed), plus per-key mask bias
            int kk = tid >> 1, d0 = (tid & 1) * 32;
            int kg = kbase + kk;
            bool ok = (kg >= 0 && kg < T_);
            const float* ks = kb + (size_t)(ok ? kg : 0) * (3 * H_) + d0;
            const float* vs = vb + (size_t)(ok ? kg : 0) * (3 * H_) + d0;
#pragma unroll
            for (int i = 0; i < 32; ++i) {
                Ks[kk * 66 + d0 + i] = ok ? f2bf(ks[i]) : (unsigned short)0;
                Vt[(d0 + i) * 66 + kk] = ok ? f2bf(vs[i]) : (unsigned short)0;
            }
            if (tid < 64) {
                int kg2 = kbase + tid;
                float m = 0.f;
                if (kg2 < 0 || kg2 >= T_ || kpm[b * T_ + kg2]) m = -1e30f;
                kAdd[tid] = m;
            }
        }
        __syncthreads();

        // S = Q @ K^T with masking written to LDS
        int llo = lane & 15, lhi = lane >> 4;
#pragma unroll
        for (int ni = 0; ni < 4; ++ni) {
            f32x8 sfrag = f32x8{0.f,0.f,0.f,0.f,0.f,0.f,0.f,0.f};
            FragU a, bb;
            ld_frag(Qs, 66, wq, 0, a);  ld_frag(Ks, 66, ni * 16, 0, bb);
            sfrag = wmma_bf16(a, bb, sfrag);
            ld_frag(Qs, 66, wq, 32, a); ld_frag(Ks, 66, ni * 16, 32, bb);
            sfrag = wmma_bf16(a, bb, sfrag);
            int kk = ni * 16 + llo;
            int kg = kbase + kk;
#pragma unroll
            for (int vr = 0; vr < 8; ++vr) {
                int qr = wq + vr + lhi * 8;
                int qg = q0 + qr;
                float sv = sfrag[vr] + kAdd[kk];
                int dq = qg - kg; if (dq < 0) dq = -dq;
                if (dq > WINDOW_) sv = -1e30f;
                if (qg == kg) sv = sfrag[vr];  // diagonal always attends
                Sf[qr * 66 + kk] = sv;
            }
        }
        __syncthreads();

        // online softmax per query row
        if (tid < 64) {
            int r = tid;
            float mOld = sM[r], tm = -1e30f;
            for (int j = 0; j < 64; ++j) tm = fmaxf(tm, Sf[r * 66 + j]);
            float mNew = fmaxf(mOld, tm);
            float alpha = __expf(mOld - mNew);
            bool dead = (mNew < -1e29f);
            float ts = 0.f;
            for (int j = 0; j < 64; ++j) {
                float p = dead ? 0.f : __expf(Sf[r * 66 + j] - mNew);
                ts += p;
                Ps[r * 66 + j] = f2bf(p);
            }
            sL[r] = sL[r] * alpha + ts;
            sM[r] = mNew;
            sA[r] = alpha;
        }
        __syncthreads();

        {   // rescale O and accumulate P @ V
            float al[8];
#pragma unroll
            for (int vr = 0; vr < 8; ++vr) al[vr] = sA[wq + vr + lhi * 8];
#pragma unroll
            for (int ni = 0; ni < 4; ++ni) {
#pragma unroll
                for (int vr = 0; vr < 8; ++vr) o[ni][vr] *= al[vr];
                FragU a, bb;
                ld_frag(Ps, 66, wq, 0, a);  ld_frag(Vt, 66, ni * 16, 0, bb);
                o[ni] = wmma_bf16(a, bb, o[ni]);
                ld_frag(Ps, 66, wq, 32, a); ld_frag(Vt, 66, ni * 16, 32, bb);
                o[ni] = wmma_bf16(a, bb, o[ni]);
            }
        }
        __syncthreads();
    }

    int llo = lane & 15, lhi = lane >> 4;
#pragma unroll
    for (int ni = 0; ni < 4; ++ni) {
#pragma unroll
        for (int vr = 0; vr < 8; ++vr) {
            int qr = wq + vr + lhi * 8;
            int qg = q0 + qr;
            float l = sL[qr];
            float inv = (l > 0.f) ? (1.f / l) : 0.f;
            outO[((size_t)b * T_ + qg) * H_ + h * DH_ + ni * 16 + llo] =
                o[ni][vr] * inv;
        }
    }
}

// ---------------- small support kernels ----------------
// convert fp32 [K,N] weight -> bf16 TRANSPOSED [N,K]
__global__ void cvtT_kernel(const float* __restrict__ src,
                            unsigned short* __restrict__ dst, int Kd, int Nd) {
    int idx = blockIdx.x * 256 + threadIdx.x;
    if (idx >= Kd * Nd) return;
    int k = idx / Nd, n = idx % Nd;          // coalesced read
    dst[(size_t)n * Kd + k] = f2bf(src[idx]); // strided write (once per launch)
}

__global__ void copy_f32_kernel(const float* __restrict__ s, float* __restrict__ d, int n) {
    int i = blockIdx.x * 256 + threadIdx.x;
    if (i < n) d[i] = s[i];
}

// gb[b, c] = s[b,:] @ aff_w[:, c] + aff_b[c],  c in [0, 2H)
__global__ void gb_kernel(const float* __restrict__ s, const float* __restrict__ affw,
                          const float* __restrict__ affb, float* __restrict__ gb) {
    int idx = blockIdx.x * 256 + threadIdx.x;   // B * 2H
    int b = idx >> 10, c = idx & 1023;
    float acc = affb[c];
    const float* sr = s + b * SD_;
    for (int i = 0; i < SD_; ++i) acc += sr[i] * affw[i * (2 * H_) + c];
    gb[idx] = acc;
}

// per-row LayerNorm + FiLM: out = (1+gamma)*LN(x) + beta
__global__ __launch_bounds__(256)
void saln_kernel(const float* __restrict__ x, const float* __restrict__ gb,
                 const float* __restrict__ lnw, const float* __restrict__ lnb,
                 float* __restrict__ out) {
    __shared__ float rs[256], rq[256];
    int row = blockIdx.x, t = threadIdx.x;
    int b = row / T_;
    const float* xr = x + (size_t)row * H_;
    float v0 = xr[t], v1 = xr[t + 256];
    rs[t] = v0 + v1; rq[t] = v0 * v0 + v1 * v1;
    __syncthreads();
    for (int o = 128; o > 0; o >>= 1) {
        if (t < o) { rs[t] += rs[t + o]; rq[t] += rq[t + o]; }
        __syncthreads();
    }
    float mean = rs[0] / H_;
    float var  = rq[0] / H_ - mean * mean;
    float inv  = rsqrtf(var + EPS_);
    const float* g = gb + b * (2 * H_);
    float* orow = out + (size_t)row * H_;
    for (int c = t; c < H_; c += 256) {
        float y = (xr[c] - mean) * inv * lnw[c] + lnb[c];
        orow[c] = (1.f + g[c]) * y + g[H_ + c];
    }
}

__global__ __launch_bounds__(256)
void final_ln_kernel(const float* __restrict__ x, const float* __restrict__ w,
                     const float* __restrict__ bb, float* __restrict__ out) {
    __shared__ float rs[256], rq[256];
    int row = blockIdx.x, t = threadIdx.x;
    const float* xr = x + (size_t)row * H_;
    float v0 = xr[t], v1 = xr[t + 256];
    rs[t] = v0 + v1; rq[t] = v0 * v0 + v1 * v1;
    __syncthreads();
    for (int o = 128; o > 0; o >>= 1) {
        if (t < o) { rs[t] += rs[t + o]; rq[t] += rq[t + o]; }
        __syncthreads();
    }
    float mean = rs[0] / H_;
    float var  = rq[0] / H_ - mean * mean;
    float inv  = rsqrtf(var + EPS_);
    float* orow = out + (size_t)row * H_;
    for (int c = t; c < H_; c += 256)
        orow[c] = (xr[c] - mean) * inv * w[c] + bb[c];
}

__global__ void glu_kernel(const float* __restrict__ u, float* __restrict__ out) {
    int idx = blockIdx.x * 256 + threadIdx.x;
    if (idx >= BTH_) return;
    int row = idx / H_, c = idx % H_;
    float a = u[(size_t)row * (2 * H_) + c];
    float g = u[(size_t)row * (2 * H_) + H_ + c];
    out[idx] = a * (1.f / (1.f + __expf(-g)));
}

__global__ void dwconv_kernel(const float* __restrict__ g, const float* __restrict__ w,
                              const float* __restrict__ bias, float* __restrict__ out) {
    int idx = blockIdx.x * 256 + threadIdx.x;
    if (idx >= BTH_) return;
    int c = idx % H_;
    int t = (idx / H_) % T_;
    int b = idx / (H_ * T_);
    float acc = bias[c];
    const float* gbp = g + (size_t)b * T_ * H_;
#pragma unroll
    for (int k = 0; k < KCONV_; ++k) {
        int tt = t + k - PADC_;
        if (tt >= 0 && tt < T_) acc += gbp[(size_t)tt * H_ + c] * w[k * H_ + c];
    }
    out[idx] = acc;
}

__global__ __launch_bounds__(256)
void gn_reduce_kernel(const float* __restrict__ g, float* __restrict__ stats) {
    __shared__ float rs[256], rq[256];
    int b = blockIdx.x, t = threadIdx.x;
    const float* p = g + (size_t)b * T_ * H_;
    float s = 0.f, q = 0.f;
    for (int i = t; i < T_ * H_; i += 256) { float v = p[i]; s += v; q += v * v; }
    rs[t] = s; rq[t] = q;
    __syncthreads();
    for (int o = 128; o > 0; o >>= 1) {
        if (t < o) { rs[t] += rs[t + o]; rq[t] += rq[t + o]; }
        __syncthreads();
    }
    if (t == 0) {
        float m = rs[0] / (float)(T_ * H_);
        stats[2 * b] = m;
        stats[2 * b + 1] = rq[0] / (float)(T_ * H_) - m * m;
    }
}

__global__ void gn_apply_kernel(float* __restrict__ g, const float* __restrict__ stats,
                                const float* __restrict__ w, const float* __restrict__ bb) {
    int idx = blockIdx.x * 256 + threadIdx.x;
    if (idx >= BTH_) return;
    int c = idx % H_;
    int b = idx / (T_ * H_);
    float m = stats[2 * b], v = stats[2 * b + 1];
    float y = (g[idx] - m) * rsqrtf(v + EPS_) * w[c] + bb[c];
    g[idx] = y / (1.f + __expf(-y));   // SiLU
}

// ---------------- host orchestration ----------------
extern "C" void kernel_launch(void* const* d_in, const int* in_sizes, int n_in,
                              void* d_out, int out_size, void* d_ws, size_t ws_size,
                              hipStream_t stream) {
    (void)in_sizes; (void)n_in; (void)out_size; (void)ws_size;

    const float* x   = (const float*)d_in[0];
    const float* s   = (const float*)d_in[1];
    const unsigned char* kpm = (const unsigned char*)d_in[2];
    // d_in[3] = attn_mask (recomputed analytically on device)

    const float* ln1w = (const float*)d_in[4];
    const float* ln1b = (const float*)d_in[5];
    const float* aw1  = (const float*)d_in[6];
    const float* ab1  = (const float*)d_in[7];
    const float* w1a  = (const float*)d_in[8];
    const float* b1a  = (const float*)d_in[9];
    const float* w2a  = (const float*)d_in[10];
    const float* b2a  = (const float*)d_in[11];
    const float* ln2w = (const float*)d_in[12];
    const float* ln2b = (const float*)d_in[13];
    const float* aw2  = (const float*)d_in[14];
    const float* ab2  = (const float*)d_in[15];
    const float* wqkv = (const float*)d_in[16];
    const float* bqkv = (const float*)d_in[17];
    const float* wo   = (const float*)d_in[18];
    const float* bo   = (const float*)d_in[19];
    const float* ln3w = (const float*)d_in[20];
    const float* ln3b = (const float*)d_in[21];
    const float* aw3  = (const float*)d_in[22];
    const float* ab3  = (const float*)d_in[23];
    const float* pw1w = (const float*)d_in[24];
    const float* pw1b = (const float*)d_in[25];
    const float* dww  = (const float*)d_in[26];
    const float* dwb  = (const float*)d_in[27];
    const float* gnw  = (const float*)d_in[28];
    const float* gnb  = (const float*)d_in[29];
    const float* pw2w = (const float*)d_in[30];
    const float* pw2b = (const float*)d_in[31];
    const float* ln4w = (const float*)d_in[32];
    const float* ln4b = (const float*)d_in[33];
    const float* aw4  = (const float*)d_in[34];
    const float* ab4  = (const float*)d_in[35];
    const float* w1b  = (const float*)d_in[36];
    const float* b1b  = (const float*)d_in[37];
    const float* w2b  = (const float*)d_in[38];
    const float* b2b  = (const float*)d_in[39];
    const float* flnw = (const float*)d_in[40];
    const float* flnb = (const float*)d_in[41];

    // workspace carve-up
    char* wsb = (char*)d_ws;
    size_t off = 0;
    auto take = [&](size_t bytes) -> void* {
        void* p = wsb + off;
        off = (off + bytes + 255) & ~(size_t)255;
        return p;
    };
    float* res  = (float*)take((size_t)BTH_ * 4);
    float* sal  = (float*)take((size_t)BTH_ * 4);
    float* g1   = (float*)take((size_t)BTH_ * 4);
    float* g2   = (float*)take((size_t)BTH_ * 4);
    float* hbuf = (float*)take((size_t)B_ * T_ * INNER_ * 4);   // 2048 cols max
    float* gbb  = (float*)take((size_t)B_ * 2 * H_ * 4);
    float* stats = (float*)take(2 * B_ * 4);
    unsigned short* w1a_bf  = (unsigned short*)take((size_t)H_ * INNER_ * 2);
    unsigned short* w2a_bf  = (unsigned short*)take((size_t)INNER_ * H_ * 2);
    unsigned short* wqkv_bf = (unsigned short*)take((size_t)H_ * 3 * H_ * 2);
    unsigned short* wo_bf   = (unsigned short*)take((size_t)H_ * H_ * 2);
    unsigned short* pw1_bf  = (unsigned short*)take((size_t)H_ * 2 * H_ * 2);
    unsigned short* pw2_bf  = (unsigned short*)take((size_t)H_ * H_ * 2);
    unsigned short* w1b_bf  = (unsigned short*)take((size_t)H_ * INNER_ * 2);
    unsigned short* w2b_bf  = (unsigned short*)take((size_t)INNER_ * H_ * 2);

    // weights converted to bf16 AND transposed to [N][K]
    auto CVT = [&](const float* src, unsigned short* dst, int Kd, int Nd) {
        int n = Kd * Nd;
        cvtT_kernel<<<(n + 255) / 256, 256, 0, stream>>>(src, dst, Kd, Nd);
    };
    CVT(w1a,  w1a_bf,  H_,     INNER_);
    CVT(w2a,  w2a_bf,  INNER_, H_);
    CVT(wqkv, wqkv_bf, H_,     3 * H_);
    CVT(wo,   wo_bf,   H_,     H_);
    CVT(pw1w, pw1_bf,  H_,     2 * H_);
    CVT(pw2w, pw2_bf,  H_,     H_);
    CVT(w1b,  w1b_bf,  H_,     INNER_);
    CVT(w2b,  w2b_bf,  INNER_, H_);

    copy_f32_kernel<<<BTH_ / 256, 256, 0, stream>>>(x, res, BTH_);

    auto GEMM = [&](const float* A, const unsigned short* Bt, const float* bias,
                    float* out, const float* resid, const unsigned char* kp,
                    int M, int N, int K, int mode, float sc) {
        dim3 grid(N / 128, M / 128);
        gemm_wmma<<<grid, 256, 0, stream>>>(A, Bt, bias, out, resid, kp,
                                            M, N, K, mode, sc);
    };

    // ---- FFN 1: x += 0.5 * FFN(SALN(x)) ----
    gb_kernel<<<(B_ * 2 * H_) / 256, 256, 0, stream>>>(s, aw1, ab1, gbb);
    saln_kernel<<<MROWS_, 256, 0, stream>>>(res, gbb, ln1w, ln1b, sal);
    GEMM(sal, w1a_bf, b1a, hbuf, nullptr, nullptr, MROWS_, INNER_, H_, EPI_SILU, 0.f);
    GEMM(hbuf, w2a_bf, b2a, res, res, nullptr, MROWS_, H_, INNER_, EPI_RES, 0.5f);

    // ---- MHSA: x += MHSA(SALN(x)) (banded window, zero padded rows) ----
    gb_kernel<<<(B_ * 2 * H_) / 256, 256, 0, stream>>>(s, aw2, ab2, gbb);
    saln_kernel<<<MROWS_, 256, 0, stream>>>(res, gbb, ln2w, ln2b, sal);
    GEMM(sal, wqkv_bf, bqkv, hbuf, nullptr, nullptr, MROWS_, 3 * H_, H_, EPI_BIAS, 0.f);
    attn_wmma<<<dim3(T_ / 64, B_ * NH_), 128, 0, stream>>>(hbuf, kpm, sal);
    GEMM(sal, wo_bf, bo, res, res, kpm, MROWS_, H_, H_, EPI_RESMASK, 1.f);

    // ---- Conv module: x += Conv(SALN(x)) ----
    gb_kernel<<<(B_ * 2 * H_) / 256, 256, 0, stream>>>(s, aw3, ab3, gbb);
    saln_kernel<<<MROWS_, 256, 0, stream>>>(res, gbb, ln3w, ln3b, sal);
    GEMM(sal, pw1_bf, pw1b, hbuf, nullptr, nullptr, MROWS_, 2 * H_, H_, EPI_BIAS, 0.f);
    glu_kernel<<<BTH_ / 256, 256, 0, stream>>>(hbuf, g1);
    dwconv_kernel<<<BTH_ / 256, 256, 0, stream>>>(g1, dww, dwb, g2);
    gn_reduce_kernel<<<B_, 256, 0, stream>>>(g2, stats);
    gn_apply_kernel<<<BTH_ / 256, 256, 0, stream>>>(g2, stats, gnw, gnb);
    GEMM(g2, pw2_bf, pw2b, res, res, nullptr, MROWS_, H_, H_, EPI_RES, 1.f);

    // ---- FFN 2: x += 0.5 * FFN(SALN(x)) ----
    gb_kernel<<<(B_ * 2 * H_) / 256, 256, 0, stream>>>(s, aw4, ab4, gbb);
    saln_kernel<<<MROWS_, 256, 0, stream>>>(res, gbb, ln4w, ln4b, sal);
    GEMM(sal, w1b_bf, b1b, hbuf, nullptr, nullptr, MROWS_, INNER_, H_, EPI_SILU, 0.f);
    GEMM(hbuf, w2b_bf, b2b, res, res, nullptr, MROWS_, H_, INNER_, EPI_RES, 0.5f);

    // ---- final LayerNorm ----
    final_ln_kernel<<<MROWS_, 256, 0, stream>>>(res, flnw, flnb, (float*)d_out);
}
